// DecoderWithAttention_15075335209366
// MI455X (gfx1250) — matrix operand order repaired and checked by reading, not compiled, not run on hardware
//
#include <hip/hip_runtime.h>
#include <hip/hip_bf16.h>
#include <cstdint>
#include <cstddef>

// Problem constants (match reference)
#define B_  32
#define T_  64
#define S_  64
#define H_  1024
#define E_  512
#define V_  32000
#define L_  4
#define TS_ 63   // T-1 decode steps

typedef __bf16        v16bf __attribute__((ext_vector_type(16)));
typedef float         v8f   __attribute__((ext_vector_type(8)));
typedef unsigned int  v4u   __attribute__((ext_vector_type(4)));
typedef unsigned int  v8u   __attribute__((ext_vector_type(8)));

__device__ __forceinline__ uint16_t f2bf(float x) {
    uint32_t u = __float_as_uint(x);
    u += 0x7FFFu + ((u >> 16) & 1u);      // round-to-nearest-even
    return (uint16_t)(u >> 16);
}
__device__ __forceinline__ float bf2f(uint16_t h) {
    return __uint_as_float(((uint32_t)h) << 16);
}
__device__ __forceinline__ float sigm(float x) { return 1.0f / (1.0f + expf(-x)); }

// ---------------------------------------------------------------------------
// Weight packing: fp32 row-major [N,K] (optionally two K-concatenated sources,
// computing A @ W^T) -> bf16 WMMA B-fragment order.
// Fragment index = kb * (N/16) + n_tile (k-major: the NT consecutive n-tiles a
// wave works on are one contiguous block per k-step).
// Fragment = 256 dwords: dword[(frag*32+lane)*8 + i] holds
//   K = kb*32 + (lane>>4)*16 + 2i..+1   for column N = n_tile*16 + lane%16.
// ginter != 0 (LSTM): packed column n' maps to source row (n'%4)*ginter + n'/4
// so one n-tile holds 4 hidden units x 4 gates (i,f,g,o interleaved).
// ---------------------------------------------------------------------------
__global__ __launch_bounds__(256) void pack_weight_bf16(
    const float* __restrict__ W0, int K0,
    const float* __restrict__ W1, int K1,
    uint32_t* __restrict__ out, int numN, int ginter, long long total)
{
    long long tid = (long long)blockIdx.x * 256 + threadIdx.x;
    if (tid >= total) return;
    int i    = (int)(tid & 7);
    int lane = (int)((tid >> 3) & 31);
    long long frag = tid >> 8;
    int n_tile = (int)(frag % numN);
    int kb     = (int)(frag / numN);
    int n = n_tile * 16 + (lane & 15);
    if (ginter) n = (n & 3) * ginter + (n >> 2);      // gate-interleaved column
    int k = kb * 32 + (lane >> 4) * 16 + i * 2;
    float f0, f1;
    if (k < K0) {
        f0 = W0[(size_t)n * K0 + k];
        f1 = W0[(size_t)n * K0 + k + 1];
    } else {
        int kk = k - K0;
        f0 = W1[(size_t)n * K1 + kk];
        f1 = W1[(size_t)n * K1 + kk + 1];
    }
    out[tid] = (uint32_t)f2bf(f0) | ((uint32_t)f2bf(f1) << 16);
}

// ---------------------------------------------------------------------------
// bf16 WMMA GEMM: C[32,N] = A[32,K] @ Wpacked^T + bias.
// 8 waves/block: wave -> (m_tile = w&1, NT consecutive n-tiles per wave).
// Branch-free ping-pong software pipeline, two k-steps per loop iteration:
//   load set1(kb+1); wmma(set0); load set0(kb+2); wmma(set1)
// REQUIRES numKblk even; final iteration's lookahead load overshoots by one
// k-step (weight buffers are padded by numN*1KB, A buffers by 256B).
// MODE 0: store fp32 (acc+bias).
// MODE 1: store bf16 tanh(acc+bias).
// MODE 2: fused LSTM cell epilogue on gate-interleaved columns:
//         col n' -> hidden j = n'>>2, gate = n'&3 (i,f,g,o); the 4 gates of
//         (row,j) live in 4 adjacent lanes of the same lane-half -> __shfl
//         gather, then c_new = sig(f)*c + sig(i)*tanh(g),
//         h = sig(o)*tanh(c_new) written bf16 to out1/out2, c updated.
// ---------------------------------------------------------------------------
template <int NT, int MODE>
__global__ __launch_bounds__(256) void gemm_bf16_wmma(
    const uint16_t* __restrict__ A, int K,
    const uint32_t* __restrict__ Bp,
    const float*    __restrict__ bias,
    float*          __restrict__ C,
    uint16_t*       __restrict__ Cbf, int N,
    float*          __restrict__ cst,            // MODE 2: cell state [32,H]
    uint16_t*       __restrict__ out1, int s1,   // MODE 2: h dest 1 (offset 0)
    uint16_t*       __restrict__ out2, int s2, int o2)  // MODE 2: h dest 2
{
    const int wave   = threadIdx.x >> 5;
    const int lane   = threadIdx.x & 31;
    const int m_tile = wave & 1;
    const int numN   = N >> 4;
    const int nt0    = (blockIdx.x * 4 + (wave >> 1)) * NT;
    if (nt0 >= numN) return;                       // wave-uniform guard
    const int numKblk = K >> 5;                    // even for all call sites
    const int khalf   = lane >> 4;
    const int mrow    = m_tile * 16 + (lane & 15);

    const uint16_t* ap = A + (size_t)mrow * K + khalf * 16;
    const uint32_t* bp = Bp + ((size_t)nt0 * 32 + lane) * 8;
    const size_t bstride = (size_t)numN * 256;     // dwords per k-step

    v8f acc[NT];
#pragma unroll
    for (int j = 0; j < NT; ++j) acc[j] = (v8f){};

    v4u al0, ah0, al1, ah1;
    v8u b0[NT], b1[NT];

    // prologue: set0 <- kb = 0
    al0 = *(const v4u*)(ap);
    ah0 = *(const v4u*)(ap + 8);
#pragma unroll
    for (int j = 0; j < NT; ++j) b0[j] = *(const v8u*)(bp + (size_t)j * 256);

    for (int kb = 0; kb < numKblk; kb += 2) {
        // set1 <- kb+1
        ap += 32; bp += bstride;
        al1 = *(const v4u*)(ap);
        ah1 = *(const v4u*)(ap + 8);
#pragma unroll
        for (int j = 0; j < NT; ++j) b1[j] = *(const v8u*)(bp + (size_t)j * 256);

        {   // wmma on set0 (kb)
            v8u afr;
            afr[0] = al0[0]; afr[1] = al0[1]; afr[2] = al0[2]; afr[3] = al0[3];
            afr[4] = ah0[0]; afr[5] = ah0[1]; afr[6] = ah0[2]; afr[7] = ah0[3];
            const v16bf av = __builtin_bit_cast(v16bf, afr);
#pragma unroll
            for (int j = 0; j < NT; ++j)
                acc[j] = __builtin_amdgcn_wmma_f32_16x16x32_bf16(
                    false, av, false, __builtin_bit_cast(v16bf, b0[j]),
                    (short)0, acc[j], false, false);
        }

        // set0 <- kb+2 (overshoots on last iteration; buffers padded)
        ap += 32; bp += bstride;
        al0 = *(const v4u*)(ap);
        ah0 = *(const v4u*)(ap + 8);
#pragma unroll
        for (int j = 0; j < NT; ++j) b0[j] = *(const v8u*)(bp + (size_t)j * 256);

        {   // wmma on set1 (kb+1)
            v8u afr;
            afr[0] = al1[0]; afr[1] = al1[1]; afr[2] = al1[2]; afr[3] = al1[3];
            afr[4] = ah1[0]; afr[5] = ah1[1]; afr[6] = ah1[2]; afr[7] = ah1[3];
            const v16bf av = __builtin_bit_cast(v16bf, afr);
#pragma unroll
            for (int j = 0; j < NT; ++j)
                acc[j] = __builtin_amdgcn_wmma_f32_16x16x32_bf16(
                    false, av, false, __builtin_bit_cast(v16bf, b1[j]),
                    (short)0, acc[j], false, false);
        }
    }

    // C layout: lane L, VGPR r -> row m_tile*16 + khalf*8 + r, col tile*16 + L%16
    const int row0  = m_tile * 16 + khalf * 8;
    const int col16 = lane & 15;
#pragma unroll
    for (int j = 0; j < NT; ++j) {
        const int n  = (nt0 + j) * 16 + col16;
        const float bv = bias ? bias[n] : 0.0f;
        if (MODE == 2) {
            // gather lanes for this row's (i,f,g,o) of hidden jj
            const int baseL = (lane & 16) | (col16 & 12);
            const int jj    = (nt0 + j) * 4 + (col16 >> 2);
#pragma unroll
            for (int r = 0; r < 8; ++r) {
                const float gval = acc[j][r] + bv;
                const float gi = __shfl(gval, baseL + 0, 32);
                const float gf = __shfl(gval, baseL + 1, 32);
                const float gg = __shfl(gval, baseL + 2, 32);
                const float go = __shfl(gval, baseL + 3, 32);
                if ((col16 & 3) == 0) {
                    const int row = row0 + r;
                    const size_t ci = (size_t)row * H_ + jj;
                    const float cn = sigm(gf) * cst[ci] + sigm(gi) * tanhf(gg);
                    cst[ci] = cn;
                    const uint16_t hb = f2bf(sigm(go) * tanhf(cn));
                    out1[(size_t)row * s1 + jj]      = hb;
                    out2[(size_t)row * s2 + o2 + jj] = hb;
                }
            }
        } else {
#pragma unroll
            for (int r = 0; r < 8; ++r) {
                const float v = acc[j][r] + bv;
                if (MODE == 0) C[(size_t)(row0 + r) * N + n] = v;
                else           Cbf[(size_t)(row0 + r) * N + n] = f2bf(tanhf(v));
            }
        }
    }
}

// ---------------------------------------------------------------------------
// Small elementwise / setup kernels
// ---------------------------------------------------------------------------
// out[n'] = a[src] + b[src]; src gate-interleave-permuted when ginter != 0
__global__ __launch_bounds__(256) void combine_bias(
    const float* __restrict__ a, const float* __restrict__ b,
    float* __restrict__ out, int n, int ginter)
{
    int i = blockIdx.x * 256 + threadIdx.x;
    if (i >= n) return;
    int src = ginter ? ((i & 3) * ginter + (i >> 2)) : i;
    out[i] = a[src] + b[src];
}

// copy h0 for one layer into its h_prev slot inside an A buffer
__global__ __launch_bounds__(256) void copy_h0_bf16(
    const float* __restrict__ h0_l, uint16_t* __restrict__ dst,
    int stride, int off)
{
    int i = blockIdx.x * 256 + threadIdx.x;
    if (i >= B_ * H_) return;
    int b = i / H_, j = i % H_;
    dst[(size_t)b * stride + off + j] = f2bf(h0_l[i]);
}

__global__ __launch_bounds__(256) void init_misc(
    const float* __restrict__ c0, float* __restrict__ c_state,
    uint16_t* __restrict__ h_tilde)
{
    int i = blockIdx.x * 256 + threadIdx.x;
    if (i < L_ * B_ * H_) c_state[i] = c0[i];
    if (i < B_ * H_)      h_tilde[i] = 0;
}

// x = [embed(tok) | h_tilde] written into layer-0 A buffer (stride 2560)
__global__ __launch_bounds__(256) void embed_concat(
    const int* __restrict__ tgt, const float* __restrict__ emb,
    const uint16_t* __restrict__ h_tilde, uint16_t* __restrict__ A0, int t)
{
    int idx = blockIdx.x * 256 + threadIdx.x;
    if (idx >= B_ * (E_ + H_)) return;
    int b = idx / (E_ + H_);
    int k = idx % (E_ + H_);
    uint16_t v;
    if (k < E_) {
        int tok = tgt[b * T_ + t];
        v = f2bf(emb[(size_t)tok * E_ + k]);
    } else {
        v = h_tilde[b * H_ + (k - E_)];
    }
    A0[(size_t)b * (E_ + H_ + H_) + k] = v;
}

// ---------------------------------------------------------------------------
// Fused attention per batch row: scores -> mask -> softmax -> context;
// writes attn probs to output and [ctx | h_t] bf16 as next GEMM input.
// ---------------------------------------------------------------------------
__global__ __launch_bounds__(256) void attention_kernel(
    const float* __restrict__ enc,        // [B,S,H]
    const int*   __restrict__ mask,       // [B,S]
    const uint16_t* __restrict__ h_t,     // [B,H] bf16
    uint16_t* __restrict__ Ac,            // [B, 2H] bf16 out: [ctx | h_t]
    float* __restrict__ attn_out, int t)  // attn section of d_out
{
    const int b = blockIdx.x;
    __shared__ float sc[S_];
    __shared__ float hsh[H_];
    __shared__ float invsum;

    for (int i = threadIdx.x; i < H_; i += 256)
        hsh[i] = bf2f(h_t[b * H_ + i]);
    __syncthreads();

    const int wave = threadIdx.x >> 5;
    const int lane = threadIdx.x & 31;
    for (int s = wave * 8; s < wave * 8 + 8; ++s) {
        const float* e = enc + ((size_t)b * S_ + s) * H_;
        float p = 0.0f;
        for (int i = lane; i < H_; i += 32) p += e[i] * hsh[i];
        for (int off = 16; off > 0; off >>= 1) p += __shfl_down(p, off, 32);
        if (lane == 0) sc[s] = p;
    }
    __syncthreads();

    if (threadIdx.x == 0) {
        const float NEG = -3.402823466e38f;
        float mx = NEG;
        for (int s = 0; s < S_; ++s) {
            float v = (mask[b * S_ + s] == 0) ? NEG : sc[s];
            sc[s] = v;
            if (v > mx) mx = v;
        }
        float sum = 0.0f;
        for (int s = 0; s < S_; ++s) { float e2 = expf(sc[s] - mx); sc[s] = e2; sum += e2; }
        invsum = 1.0f / sum;
    }
    __syncthreads();

    if (threadIdx.x < S_) {
        float p = sc[threadIdx.x] * invsum;
        sc[threadIdx.x] = p;
        attn_out[((size_t)b * TS_ + t) * S_ + threadIdx.x] = p;
    }
    __syncthreads();

    for (int hh = threadIdx.x; hh < H_; hh += 256) {
        float acc = 0.0f;
        const float* e = enc + (size_t)b * S_ * H_ + hh;
        for (int s = 0; s < S_; ++s) acc += sc[s] * e[(size_t)s * H_];
        Ac[(size_t)b * 2 * H_ + hh]      = f2bf(acc);
        Ac[(size_t)b * 2 * H_ + H_ + hh] = f2bf(hsh[hh]);
    }
}

// ---------------------------------------------------------------------------
// Row log-softmax over V=32000; one block per batch row.
// ---------------------------------------------------------------------------
__global__ __launch_bounds__(256) void logsoftmax_kernel(
    const float* __restrict__ logits, float* __restrict__ out, int t)
{
    const int b = blockIdx.x;
    const float* row = logits + (size_t)b * V_;
    __shared__ float sdata[256];

    float mx = -3.402823466e38f;
    for (int i = threadIdx.x; i < V_; i += 256) mx = fmaxf(mx, row[i]);
    sdata[threadIdx.x] = mx;
    __syncthreads();
    for (int off = 128; off > 0; off >>= 1) {
        if (threadIdx.x < off)
            sdata[threadIdx.x] = fmaxf(sdata[threadIdx.x], sdata[threadIdx.x + off]);
        __syncthreads();
    }
    mx = sdata[0];
    __syncthreads();

    float sum = 0.0f;
    for (int i = threadIdx.x; i < V_; i += 256) sum += expf(row[i] - mx);
    sdata[threadIdx.x] = sum;
    __syncthreads();
    for (int off = 128; off > 0; off >>= 1) {
        if (threadIdx.x < off) sdata[threadIdx.x] += sdata[threadIdx.x + off];
        __syncthreads();
    }
    const float lse = mx + logf(sdata[0]);

    float* o = out + ((size_t)b * TS_ + t) * V_;
    for (int i = threadIdx.x; i < V_; i += 256) o[i] = row[i] - lse;
}

// ---------------------------------------------------------------------------
// Host driver
// ---------------------------------------------------------------------------
extern "C" void kernel_launch(void* const* d_in, const int* in_sizes, int n_in,
                              void* d_out, int out_size, void* d_ws, size_t ws_size,
                              hipStream_t stream)
{
    (void)in_sizes; (void)n_in; (void)out_size; (void)ws_size;

    // setup_inputs() order (params dict flattened in insertion order)
    const int*   tgt  = (const int*)d_in[0];
    const float* enc  = (const float*)d_in[1];
    const int*   mask = (const int*)d_in[2];
    const float* h0   = (const float*)d_in[3];
    const float* c0   = (const float*)d_in[4];
    const float* emb  = (const float*)d_in[5];
    const float* Wih[L_] = {(const float*)d_in[6],  (const float*)d_in[7],
                            (const float*)d_in[8],  (const float*)d_in[9]};
    const float* Whh[L_] = {(const float*)d_in[10], (const float*)d_in[11],
                            (const float*)d_in[12], (const float*)d_in[13]};
    const float* bih[L_] = {(const float*)d_in[14], (const float*)d_in[15],
                            (const float*)d_in[16], (const float*)d_in[17]};
    const float* bhh[L_] = {(const float*)d_in[18], (const float*)d_in[19],
                            (const float*)d_in[20], (const float*)d_in[21]};
    const float* Wc = (const float*)d_in[22];
    const float* bc = (const float*)d_in[23];
    const float* Wo = (const float*)d_in[24];
    const float* bo = (const float*)d_in[25];

    // ---- workspace layout ----
    char* ws = (char*)d_ws;
    size_t off = 0;
    auto walloc = [&](size_t bytes) -> char* {
        char* p = ws + off;
        off = (off + bytes + 255) & ~(size_t)255;
        return p;
    };
    const int KL0 = E_ + H_ + H_;              // 2560, layer-0 GEMM K
    const int KLN = 2 * H_;                    // 2048, layers 1-3 / Wc GEMM K

    // packed weights; each padded by one k-step (numN fragments = numN*1KB)
    // so the pipeline's lookahead load may overshoot harmlessly.
    auto wpk_alloc = [&](int N, int K) -> uint32_t* {
        size_t frags = (size_t)(N / 16) * (K / 32);
        return (uint32_t*)walloc((frags + (size_t)(N / 16)) * 1024);
    };
    uint32_t* wpk[L_];
    wpk[0] = wpk_alloc(4 * H_, KL0);
    for (int l = 1; l < L_; ++l) wpk[l] = wpk_alloc(4 * H_, KLN);
    uint32_t* wpkC = wpk_alloc(H_, KLN);
    uint32_t* wpkO = wpk_alloc(V_, H_);

    // A buffers (+256B slack for A-side overshoot):
    uint16_t* Abuf[L_];
    Abuf[0] = (uint16_t*)walloc((size_t)B_ * KL0 * 2 + 256);         // [x(1536) | h_prev0]
    for (int l = 1; l < L_; ++l)
        Abuf[l] = (uint16_t*)walloc((size_t)B_ * KLN * 2 + 256);     // [h_{l-1} | h_prev l]
    uint16_t* Acbuf   = (uint16_t*)walloc((size_t)B_ * 2 * H_ * 2 + 256); // [ctx | h_t]
    uint16_t* h_t     = (uint16_t*)walloc((size_t)B_ * H_ * 2);
    uint16_t* h_tilde = (uint16_t*)walloc((size_t)B_ * H_ * 2 + 256);
    float*    c_state = (float*)walloc((size_t)L_ * B_ * H_ * 4);
    float*    biasL   = (float*)walloc((size_t)L_ * 4 * H_ * 4);
    float*    logits  = (float*)walloc((size_t)B_ * V_ * 4);

    float* logp_out = (float*)d_out;                         // [B, TS, V]
    float* attn_out = (float*)d_out + (size_t)B_ * TS_ * V_; // [B, TS, S]

    // ---- one-time (per launch) weight packing: fp32 -> bf16 fragments ----
    {
        long long tot = (long long)(4 * H_) * KL0 / 2;
        pack_weight_bf16<<<(unsigned)((tot + 255) / 256), 256, 0, stream>>>(
            Wih[0], E_ + H_, Whh[0], H_, wpk[0], (4 * H_) / 16, H_, tot);
    }
    for (int l = 1; l < L_; ++l) {
        long long tot = (long long)(4 * H_) * KLN / 2;
        pack_weight_bf16<<<(unsigned)((tot + 255) / 256), 256, 0, stream>>>(
            Wih[l], H_, Whh[l], H_, wpk[l], (4 * H_) / 16, H_, tot);
    }
    {
        long long tot = (long long)H_ * KLN / 2;
        pack_weight_bf16<<<(unsigned)((tot + 255) / 256), 256, 0, stream>>>(
            Wc, 2 * H_, nullptr, 0, wpkC, H_ / 16, 0, tot);
    }
    {
        long long tot = (long long)V_ * H_ / 2;
        pack_weight_bf16<<<(unsigned)((tot + 255) / 256), 256, 0, stream>>>(
            Wo, H_, nullptr, 0, wpkO, V_ / 16, 0, tot);
    }
    for (int l = 0; l < L_; ++l)
        combine_bias<<<(4 * H_ + 255) / 256, 256, 0, stream>>>(
            bih[l], bhh[l], biasL + (size_t)l * 4 * H_, 4 * H_, H_);

    // ---- state init ----
    const int grid_bh = (B_ * H_ + 255) / 256;
    copy_h0_bf16<<<grid_bh, 256, 0, stream>>>(h0 + 0 * B_ * H_, Abuf[0], KL0, E_ + H_);
    for (int l = 1; l < L_; ++l)
        copy_h0_bf16<<<grid_bh, 256, 0, stream>>>(h0 + (size_t)l * B_ * H_, Abuf[l], KLN, H_);
    init_misc<<<(L_ * B_ * H_ + 255) / 256, 256, 0, stream>>>(c0, c_state, h_tilde);

    // ---- sequential decode (strict dependence through h_tilde / h / c) ----
    for (int t = 0; t < TS_; ++t) {
        embed_concat<<<(B_ * (E_ + H_) + 255) / 256, 256, 0, stream>>>(
            tgt, emb, h_tilde, Abuf[0], t);

        for (int l = 0; l < L_; ++l) {
            const int K = (l == 0) ? KL0 : KLN;
            // fused: gates GEMM (gate-interleaved columns) + LSTM cell epilogue.
            // h -> next layer chain slot (or h_t) + this layer's h_prev slot.
            uint16_t* o1; int s1;
            if (l < L_ - 1) { o1 = Abuf[l + 1]; s1 = KLN; }
            else            { o1 = h_t;         s1 = H_;  }
            uint16_t* o2 = Abuf[l];
            int s2   = (l == 0) ? KL0 : KLN;
            int ofs2 = (l == 0) ? (E_ + H_) : H_;
            gemm_bf16_wmma<4, 2><<<(4 * H_) / 256, 256, 0, stream>>>(
                Abuf[l], K, wpk[l], biasL + (size_t)l * 4 * H_,
                nullptr, nullptr, 4 * H_,
                c_state + (size_t)l * B_ * H_, o1, s1, o2, s2, ofs2);
        }

        attention_kernel<<<B_, 256, 0, stream>>>(
            enc, mask, h_t, Acbuf, attn_out, t);

        // h_tilde = tanh([ctx | h_t] @ Wc^T + bc); N = H; NT=1 (keep waves up),
        // tanh + bf16 store fused into the GEMM epilogue.
        gemm_bf16_wmma<1, 1><<<(H_ / 16) / 4, 256, 0, stream>>>(
            Acbuf, 2 * H_, wpkC, bc, nullptr, h_tilde, H_,
            nullptr, nullptr, 0, nullptr, 0, 0);

        // logits = h_tilde @ Wo^T + bo; N = 32000 -> 125 blocks x 8 waves x 4 tiles
        gemm_bf16_wmma<4, 0><<<V_ / 256, 256, 0, stream>>>(
            h_tilde, H_, wpkO, bo, logits, nullptr, V_,
            nullptr, nullptr, 0, nullptr, 0, 0);

        logsoftmax_kernel<<<B_, 256, 0, stream>>>(logits, logp_out, t);
    }
}
